// FusedMoEModularKernel_16707422781658
// MI455X (gfx1250) — matrix operand order repaired and checked by reading, not compiled, not run on hardware
//
#include <hip/hip_runtime.h>
#include <hip/hip_bf16.h>
#include <stdint.h>

// Problem sizes (fixed by the reference)
#define M_DIM   1024
#define K_DIM   1024
#define E_NUM   8
#define N_DIM   4096
#define TOPK    2
#define I_DIM   2048                 // N/2
#define A_NUM   (M_DIM * TOPK)       // 2048 routed row-assignments
#define MAX_TILES (A_NUM / 16)       // worst case: all assignments on one expert

typedef __attribute__((ext_vector_type(16))) _Float16 v16h;
typedef __attribute__((ext_vector_type(8)))  float    v8f;

struct Routing {
    int   sorted_m[A_NUM];   // source token row per slot (expert-sorted, stable)
    float sorted_w[A_NUM];   // topk weight per slot
    int   inv_slot[A_NUM];   // assignment t -> slot (for deterministic combine)
    int   seg_start[E_NUM];
    int   seg_count[E_NUM];
};

// ---------------------------------------------------------------------------
// CDNA5 async global->LDS staging (ASYNCcnt path, CDNA5 ISA).
// Per-lane copy of 64 contiguous bytes; INST_OFFSET is added to BOTH the LDS
// destination and the global source address, so four b128 issues stream the
// chunk. LDS address operand = low 32 bits of the flat pointer (hardware maps
// flat->LDS by truncation, ISA 10.2).
// ---------------------------------------------------------------------------
__device__ __forceinline__ void async_copy_64B(uint32_t lds_addr, const float* gsrc)
{
    uint64_t ga = (uint64_t)(uintptr_t)gsrc;
    asm volatile(
        "global_load_async_to_lds_b128 %0, %1, off\n\t"
        "global_load_async_to_lds_b128 %0, %1, off offset:16\n\t"
        "global_load_async_to_lds_b128 %0, %1, off offset:32\n\t"
        "global_load_async_to_lds_b128 %0, %1, off offset:48"
        :: "v"(lds_addr), "v"(ga) : "memory");
}

__device__ __forceinline__ void wait_async_done()
{
    asm volatile("s_wait_asynccnt 0x0" ::: "memory");
}

// ---------------------------------------------------------------------------
// Routing: stable expert-sort of the 2048 (token, expert, weight) assignments.
// ---------------------------------------------------------------------------
__global__ __launch_bounds__(256) void route_kernel(
    const int* __restrict__ topk_ids, const float* __restrict__ topk_w,
    Routing* __restrict__ r)
{
    __shared__ int cnt[E_NUM];
    __shared__ int off[E_NUM];
    const int tid = threadIdx.x;
    if (tid < E_NUM) cnt[tid] = 0;
    __syncthreads();
    for (int t = tid; t < A_NUM; t += blockDim.x)
        atomicAdd(&cnt[topk_ids[t]], 1);
    __syncthreads();
    if (tid == 0) {
        int s = 0;
        for (int e = 0; e < E_NUM; ++e) {
            off[e] = s;
            r->seg_start[e] = s;
            r->seg_count[e] = cnt[e];
            s += cnt[e];
        }
    }
    __syncthreads();
    for (int t = tid; t < A_NUM; t += blockDim.x) {
        const int e = topk_ids[t];
        int pos = 0;
        for (int u = 0; u < t; ++u) pos += (topk_ids[u] == e) ? 1 : 0;
        const int slot = off[e] + pos;
        r->sorted_m[slot] = t / TOPK;
        r->sorted_w[slot] = topk_w[t];
        r->inv_slot[t]    = slot;
    }
}

// ---------------------------------------------------------------------------
// WMMA fragment index helper (wave32 16x16x32 f16, contiguous k order this code uses for A and B):
// lanes 0-15: row M=lane, K in {0..15}; lanes 16-31: K in {16..31}
// ---------------------------------------------------------------------------
__device__ __forceinline__ int a_frag_k(int lane, int j)
{
    const int kb = (lane < 16) ? 0 : 16;
    return kb + j;
}

// ---------------------------------------------------------------------------
// GEMM1 + SiLU*up fusion.
// grid = (I/64, MAX_TILES, E). Block = 256 (8 waves).
// Waves 0-3: gate tiles (w1 rows [c,c+64)); waves 4-7: matching up tiles
// (rows [c+I,c+I+64)). A tile (16 gathered token rows x 32 k, f32) is
// double-buffered in LDS via async copies issued by wave 0.
// ---------------------------------------------------------------------------
__global__ __launch_bounds__(256) void gemm1_silu_kernel(
    const float* __restrict__ hidden, const float* __restrict__ w1,
    const Routing* __restrict__ r, float* __restrict__ act)
{
    const int e    = blockIdx.z;
    const int seg0 = r->seg_start[e];
    const int cnt  = r->seg_count[e];
    const int tile = blockIdx.y;
    if (tile * 16 >= cnt) return;                    // uniform early-exit
    const int rowbase = seg0 + tile * 16;
    const int segend  = seg0 + cnt;

    const int wave = threadIdx.x >> 5;
    const int lane = threadIdx.x & 31;

    __shared__ float As[2][16][32];    // double-buffered A slab (2 x 2 KB)
    __shared__ float hbuf[16][128];    // [gate 0..63 | up 64..127]

    // Per-lane async staging descriptors (wave 0 only). Padded rows are
    // clamped to the last valid slot: they produce garbage in padded output
    // rows only, and those stores are guarded.
    const uint32_t lds0 = (uint32_t)(uintptr_t)&As[0][0][0];
    uint32_t lds_lane = 0;
    const float* gsrc = hidden;
    if (wave == 0) {
        int a  = rowbase + (lane >> 1);
        a      = (a < segend) ? a : (segend - 1);
        int mm = r->sorted_m[a];
        gsrc   = hidden + (size_t)mm * K_DIM + (lane & 1) * 16;
        lds_lane = (uint32_t)((lane >> 1) * 128 + (lane & 1) * 64);
        async_copy_64B(lds0 + lds_lane, gsrc);       // prologue: stage k0=0
    }

    const int gate_col0 = blockIdx.x * 64 + (wave & 3) * 16;
    const int n0 = gate_col0 + ((wave >= 4) ? I_DIM : 0);
    const float* __restrict__ w1row =
        w1 + (size_t)e * N_DIM * K_DIM
           + (size_t)(n0 + (lane & 15)) * K_DIM
           + ((lane < 16) ? 0 : 16);                 // B-frag K-half per lane

    v8f acc = {};
    const int arow = lane & 15;

    for (int k0 = 0; k0 < K_DIM; k0 += 32) {
        const int buf = (k0 >> 5) & 1;
        if (wave == 0) wait_async_done();            // current slab resident
        __syncthreads();                             // ...and all consumers sync'd
        if (wave == 0 && (k0 + 32) < K_DIM)          // prefetch next slab async
            async_copy_64B(lds0 + (uint32_t)((buf ^ 1) * 2048) + lds_lane,
                           gsrc + k0 + 32);

        v16h a_frag, b_frag;
#pragma unroll
        for (int j = 0; j < 16; ++j)
            a_frag[j] = (_Float16)As[buf][arow][a_frag_k(lane, j)];

        const float* bptr = w1row + k0;
        if (k0 + 32 < K_DIM) __builtin_prefetch(bptr + 32, 0, 1); // global_prefetch_b8
#pragma unroll
        for (int j = 0; j < 16; ++j)
            b_frag[j] = (_Float16)bptr[j];           // 64B contiguous per lane

        acc = __builtin_amdgcn_wmma_f32_16x16x32_f16(
                  false, a_frag, false, b_frag, (short)0, acc, false, false);
    }
    __syncthreads();

    // Scatter C tile into LDS exchange buffer (gate | up halves).
    const int cn = lane & 15;
    const int rb = (lane < 16) ? 0 : 8;
    const int colofs = (wave & 3) * 16 + cn;
    const int half = (wave >= 4) ? 64 : 0;
#pragma unroll
    for (int v = 0; v < 8; ++v)
        hbuf[rb + v][half + colofs] = acc[v];
    __syncthreads();

    // act = silu(gate) * up, written per-assignment row (guarded).
    for (int idx = threadIdx.x; idx < 16 * 64; idx += 256) {
        const int rr = idx >> 6, cc = idx & 63;
        const int a = rowbase + rr;
        if (a < segend) {
            const float g = hbuf[rr][cc];
            const float u = hbuf[rr][64 + cc];
            const float s = g / (1.0f + __expf(-g));
            act[(size_t)a * I_DIM + (size_t)blockIdx.x * 64 + cc] = s * u;
        }
    }
}

// ---------------------------------------------------------------------------
// GEMM2: out_e[a][k] = (act[a][:] . w2[e][k][:]) * topk_weight[a]
// grid = (K/128, MAX_TILES, E). 8 waves x 16 output cols each. Same async
// double-buffered A staging (act rows are contiguous slots, no gather).
// ---------------------------------------------------------------------------
__global__ __launch_bounds__(256) void gemm2_kernel(
    const float* __restrict__ act, const float* __restrict__ w2,
    const Routing* __restrict__ r, float* __restrict__ out_e)
{
    const int e    = blockIdx.z;
    const int seg0 = r->seg_start[e];
    const int cnt  = r->seg_count[e];
    const int tile = blockIdx.y;
    if (tile * 16 >= cnt) return;
    const int rowbase = seg0 + tile * 16;
    const int segend  = seg0 + cnt;

    const int wave = threadIdx.x >> 5;
    const int lane = threadIdx.x & 31;

    __shared__ float As[2][16][32];
    __shared__ float wrow[16];

    if (threadIdx.x < 16) {
        const int a = rowbase + (int)threadIdx.x;
        wrow[threadIdx.x] = (a < segend) ? r->sorted_w[a] : 0.0f;
    }

    const uint32_t lds0 = (uint32_t)(uintptr_t)&As[0][0][0];
    uint32_t lds_lane = 0;
    const float* gsrc = act;
    if (wave == 0) {
        int a = rowbase + (lane >> 1);
        a     = (a < segend) ? a : (segend - 1);
        gsrc  = act + (size_t)a * I_DIM + (lane & 1) * 16;
        lds_lane = (uint32_t)((lane >> 1) * 128 + (lane & 1) * 64);
        async_copy_64B(lds0 + lds_lane, gsrc);
    }

    const int n0 = blockIdx.x * 128 + wave * 16;     // output column base
    const float* __restrict__ w2row =
        w2 + (size_t)e * K_DIM * I_DIM
           + (size_t)(n0 + (lane & 15)) * I_DIM
           + ((lane < 16) ? 0 : 16);

    v8f acc = {};
    const int arow = lane & 15;

    for (int k0 = 0; k0 < I_DIM; k0 += 32) {
        const int buf = (k0 >> 5) & 1;
        if (wave == 0) wait_async_done();
        __syncthreads();
        if (wave == 0 && (k0 + 32) < I_DIM)
            async_copy_64B(lds0 + (uint32_t)((buf ^ 1) * 2048) + lds_lane,
                           gsrc + k0 + 32);

        v16h a_frag, b_frag;
#pragma unroll
        for (int j = 0; j < 16; ++j)
            a_frag[j] = (_Float16)As[buf][arow][a_frag_k(lane, j)];

        const float* bptr = w2row + k0;
        if (k0 + 32 < I_DIM) __builtin_prefetch(bptr + 32, 0, 1);
#pragma unroll
        for (int j = 0; j < 16; ++j)
            b_frag[j] = (_Float16)bptr[j];

        acc = __builtin_amdgcn_wmma_f32_16x16x32_f16(
                  false, a_frag, false, b_frag, (short)0, acc, false, false);
    }

    // Epilogue: scale by routing weight, store per-slot (no atomics, fully
    // deterministic).
    const int cn = lane & 15;
    const int rb = (lane < 16) ? 0 : 8;
#pragma unroll
    for (int v = 0; v < 8; ++v) {
        const int a = rowbase + rb + v;
        if (a < segend)
            out_e[(size_t)a * K_DIM + n0 + cn] = acc[v] * wrow[rb + v];
    }
}

// ---------------------------------------------------------------------------
// Deterministic combine: each token sums its two (already scaled) slots.
// Vectorized float4 (K_DIM = 1024 -> 256 float4 per row).
// ---------------------------------------------------------------------------
__global__ __launch_bounds__(256) void combine_kernel(
    const float* __restrict__ out_e, const Routing* __restrict__ r,
    float* __restrict__ out)
{
    const int idx = blockIdx.x * blockDim.x + threadIdx.x;   // over M*K/4
    if (idx >= M_DIM * (K_DIM / 4)) return;
    const int m  = idx / (K_DIM / 4);
    const int k4 = idx % (K_DIM / 4);
    const int s0 = r->inv_slot[m * TOPK + 0];
    const int s1 = r->inv_slot[m * TOPK + 1];
    const float4 a = ((const float4*)(out_e + (size_t)s0 * K_DIM))[k4];
    const float4 b = ((const float4*)(out_e + (size_t)s1 * K_DIM))[k4];
    float4 c;
    c.x = a.x + b.x; c.y = a.y + b.y; c.z = a.z + b.z; c.w = a.w + b.w;
    ((float4*)(out + (size_t)m * K_DIM))[k4] = c;
}

// ---------------------------------------------------------------------------
extern "C" void kernel_launch(void* const* d_in, const int* in_sizes, int n_in,
                              void* d_out, int out_size, void* d_ws, size_t ws_size,
                              hipStream_t stream)
{
    (void)in_sizes; (void)n_in; (void)out_size; (void)ws_size;

    const float* hidden   = (const float*)d_in[0];
    const float* w1       = (const float*)d_in[1];
    const float* w2       = (const float*)d_in[2];
    const float* topk_w   = (const float*)d_in[3];
    const int*   topk_ids = (const int*)d_in[4];
    float*       out      = (float*)d_out;

    // Workspace carve-up: act (16 MB) | out_e (8 MB) | routing (~24 KB)
    float*   act   = (float*)d_ws;
    float*   out_e = act + (size_t)A_NUM * I_DIM;
    Routing* r     = (Routing*)(out_e + (size_t)A_NUM * K_DIM);

    route_kernel<<<1, 256, 0, stream>>>(topk_ids, topk_w, r);

    dim3 g1(I_DIM / 64, MAX_TILES, E_NUM);
    gemm1_silu_kernel<<<g1, 256, 0, stream>>>(hidden, w1, r, act);

    dim3 g2(K_DIM / 128, MAX_TILES, E_NUM);
    gemm2_kernel<<<g2, 256, 0, stream>>>(act, w2, r, out_e);

    combine_kernel<<<(M_DIM * (K_DIM / 4) + 255) / 256, 256, 0, stream>>>(out_e, r, out);
}
